// GFASTKAN_Nodes_38594576122040
// MI455X (gfx1250) — matrix operand order repaired and checked by reading, not compiled, not run on hardware
//
#include <hip/hip_runtime.h>
#include <hip/hip_bf16.h>

typedef __attribute__((ext_vector_type(16))) _Float16 v16h;
typedef __attribute__((ext_vector_type(8)))  _Float16 v8h;
typedef __attribute__((ext_vector_type(4)))  _Float16 v4h;
typedef __attribute__((ext_vector_type(8)))  float    v8f;

#define NN 100000
#define FF 128
#define HH 128
#define CC 47
#define EE 1600000

// ---------------------------------------------------------------------------
// Degree / normalization:  deg = in-degree(dst) + 1 (self loop); dinv = rsqrt
// ---------------------------------------------------------------------------
__global__ void deg_init(float* deg) {
    int i = blockIdx.x * 256 + threadIdx.x;
    if (i < NN) deg[i] = 1.0f;                       // self loop
}
__global__ void deg_acc(const int* __restrict__ dst, float* deg) {
    int e = blockIdx.x * 256 + threadIdx.x;
    if (e < EE) atomicAdd(&deg[dst[e]], 1.0f);
}
__global__ void deg_finish(float* deg) {
    int i = blockIdx.x * 256 + threadIdx.x;
    if (i < NN) deg[i] = rsqrtf(deg[i]);             // deg >= 1 always
}

// ---------------------------------------------------------------------------
// Pack combined [spline | base] weights to f16 in WMMA *fragment order*:
//   Wp[colChunk][kChunk][lane][i]  (16 contiguous halfs per lane = 32B load)
//   logical source:  k = kChunk*32 + (lane<16 ? 0 : 16) + i
//                    col = colChunk*16 + (lane & 15)
//   k <  din*4 : Ws[col][k]       (spline rows, RBF features)
//   k >= din*4 : Wb[col][k-din*4] (base rows, SiLU features)
// biasc[o] = bs[o] + bb[o]  (zero-padded past dout)
// ---------------------------------------------------------------------------
template<int DIN, int DOUTP>
__global__ void prep_weights(const float* __restrict__ Ws, const float* __restrict__ Wb,
                             const float* __restrict__ bs, const float* __restrict__ bb,
                             _Float16* __restrict__ Wp, float* __restrict__ biasc, int dout) {
    const int KTOT = DIN * 5;
    const int NK   = KTOT / 32;
    int idx = blockIdx.x * 256 + threadIdx.x;
    if (idx < KTOT * DOUTP) {
        int cchunk = idx / (NK * 512);
        int rem    = idx - cchunk * (NK * 512);
        int kchunk = rem / 512;
        int within = rem - kchunk * 512;
        int lane   = within >> 4;
        int i      = within & 15;
        int k   = kchunk * 32 + ((lane < 16) ? 0 : 16) + i;
        int col = cchunk * 16 + (lane & 15);
        float v = 0.0f;
        if (col < dout) {
            if (k < DIN * 4) v = Ws[col * (DIN * 4) + k];
            else             v = Wb[col * DIN + (k - DIN * 4)];
        }
        Wp[idx] = (_Float16)v;
    }
    if (blockIdx.x == 0) {
        for (int o = threadIdx.x; o < DOUTP; o += 256)
            biasc[o] = (o < dout) ? (bs[o] + bb[o]) : 0.0f;
    }
}

// ---------------------------------------------------------------------------
// Fused FKAN layer: 16-node tile per block, 4 waves, each wave owns a 16x16
// output tile; K streamed in 32-chunks through v_wmma_f32_16x16x32_f16.
// The f16 feature matrix (4 RBF + SiLU per input) is built ONCE in LDS, so
// A-fragments are two aligned ds_load_b128 and B-fragments one coalesced
// 32B global load per lane. Layer-3 concat read from 3 segmented sources.
// ---------------------------------------------------------------------------
template<int D0, int D1, int D2, int DOUTP>
__global__ __launch_bounds__(128)
void fkan_kernel(const float* __restrict__ X0, const float* __restrict__ X1,
                 const float* __restrict__ X2,
                 const float* __restrict__ lng, const float* __restrict__ lnb,
                 const _Float16* __restrict__ Wp, const float* __restrict__ biasc,
                 float* __restrict__ out, int dout) {
    constexpr int DIN  = D0 + D1 + D2;
    constexpr int KTOT = DIN * 5;
    constexpr int NK   = KTOT / 32;

    __shared__ __align__(16) _Float16 sf[16][KTOT];   // f16 feature matrix
    __shared__ float sredA[16][8], sredB[16][8];
    __shared__ float smu[16], srs[16];

    const int tid   = threadIdx.x;
    const int lane  = tid & 31;
    const int w     = tid >> 5;
    const int node0 = blockIdx.x * 16;

    auto loadIn = [&](int n, int c) -> float {
        if (c < D0)           return X0[(long)n * D0 + c];
        else if (c < D0 + D1) return X1[(long)n * D1 + (c - D0)];
        else                  return X2[(long)n * D2 + (c - D0 - D1)];
    };

    // per-row layernorm stats: 8 threads per row
    {
        int r = tid >> 3, sub = tid & 7;
        float s = 0.0f, s2 = 0.0f;
        for (int c = sub; c < DIN; c += 8) {
            float v = loadIn(node0 + r, c);
            s += v; s2 += v * v;
        }
        sredA[r][sub] = s;
        sredB[r][sub] = s2;
    }
    __syncthreads();
    if (tid < 16) {
        float s = 0.0f, s2 = 0.0f;
#pragma unroll
        for (int j = 0; j < 8; ++j) { s += sredA[tid][j]; s2 += sredB[tid][j]; }
        float mu  = s / (float)DIN;
        float var = s2 / (float)DIN - mu * mu;
        smu[tid] = mu;
        srs[tid] = rsqrtf(var + 1e-5f);
    }
    __syncthreads();

    // build f16 feature matrix: [4 RBF of layer-normed z | SiLU(x)]
    // GRID = linspace(-2,2,4), 1/DENOM = 0.75
    for (int idx = tid; idx < 16 * DIN; idx += 128) {
        int r = idx / DIN, d = idx % DIN;
        float x = loadIn(node0 + r, d);
        float z = (x - smu[r]) * srs[r] * lng[d] + lnb[d];
        v4h rq;
#pragma unroll
        for (int gi = 0; gi < 4; ++gi) {
            float t = (z - (-2.0f + 1.33333333333f * (float)gi)) * 0.75f;
            rq[gi] = (_Float16)__expf(-t * t);
        }
        *(v4h*)&sf[r][d * 4] = rq;                               // 8B LDS store
        float sg = x * __builtin_amdgcn_rcpf(1.0f + __expf(-x)); // fast sigmoid
        sf[r][DIN * 4 + d] = (_Float16)sg;
    }
    __syncthreads();

    // WMMA: A 16x32 f16 fragment = two contiguous 8-half runs per lane
    const int arow = lane & 15;
    const int aLo  = (lane < 16) ? 0 : 16;
    const int aHi  = (lane < 16) ? 8 : 24;
    const int bcol = lane & 15;

    for (int cc = 0; cc < DOUTP; cc += 64) {
        const int colbase  = cc + w * 16;
        const int colChunk = colbase >> 4;
        const _Float16* wbase = Wp + ((long)colChunk * NK * 512) + lane * 16;
        v8f acc = {};
#pragma unroll 4
        for (int kk = 0; kk < NK; ++kk) {
            int kc = kk * 32;
            v8h alo = *(const v8h*)&sf[arow][kc + aLo];
            v8h ahi = *(const v8h*)&sf[arow][kc + aHi];
            v16h a = __builtin_shufflevector(alo, ahi,
                     0, 1, 2, 3, 4, 5, 6, 7, 8, 9, 10, 11, 12, 13, 14, 15);
            v16h b = *(const v16h*)(wbase + kk * 512);            // 32B coalesced
            acc = __builtin_amdgcn_wmma_f32_16x16x32_f16(
                      false, a, false, b, (short)0, acc, false, false);
        }
#pragma unroll
        for (int j = 0; j < 8; ++j) {
            int r  = (lane < 16) ? j : (j + 8);                   // f32 C/D layout
            int oc = colbase + bcol;
            if (oc < dout)
                out[(long)(node0 + r) * dout + oc] = acc[j] + biasc[oc];
        }
    }
}

// ---------------------------------------------------------------------------
// Aggregation: h[i] = bg + dinv[i]^2 * pre[i]  (self loop), then edge scatter
// h[dst] += dinv[src]*dinv[dst] * pre[src]   (wave per edge, f32 atomics)
// ---------------------------------------------------------------------------
__global__ void agg_init(const float* __restrict__ pre, const float* __restrict__ dinv,
                         const float* __restrict__ bg, float* __restrict__ h, int dout) {
    long idx = (long)blockIdx.x * 256 + threadIdx.x;
    long total = (long)NN * dout;
    if (idx < total) {
        int i = (int)(idx / dout);
        int f = (int)(idx % dout);
        float di = dinv[i];
        h[idx] = bg[f] + di * di * pre[idx];
    }
}

__global__ __launch_bounds__(256)
void agg_edges(const int* __restrict__ src, const int* __restrict__ dst,
               const float* __restrict__ dinv, const float* __restrict__ pre,
               float* __restrict__ h, int dout) {
    int wid  = (int)((blockIdx.x * 256 + threadIdx.x) >> 5);   // wave per edge
    int lane = threadIdx.x & 31;
    if (wid >= EE) return;
    int s = src[wid], d = dst[wid];
    float wgt = dinv[s] * dinv[d];
    const float* ps = pre + (long)s * dout;
    float*       hd = h   + (long)d * dout;
    if ((dout & 3) == 0) {
        const float4* p4 = (const float4*)ps;
        int n4 = dout >> 2;
        for (int f = lane; f < n4; f += 32) {
            float4 v = p4[f];
            float* o = hd + f * 4;
            atomicAdd(o + 0, wgt * v.x);
            atomicAdd(o + 1, wgt * v.y);
            atomicAdd(o + 2, wgt * v.z);
            atomicAdd(o + 3, wgt * v.w);
        }
    } else {
        for (int f = lane; f < dout; f += 32)
            atomicAdd(hd + f, wgt * ps[f]);
    }
}

// ---------------------------------------------------------------------------
// Launch
// ---------------------------------------------------------------------------
extern "C" void kernel_launch(void* const* d_in, const int* in_sizes, int n_in,
                              void* d_out, int out_size, void* d_ws, size_t ws_size,
                              hipStream_t stream) {
    (void)in_sizes; (void)n_in; (void)out_size; (void)ws_size;

    const float* x   = (const float*)d_in[0];
    const int*   ei  = (const int*)d_in[1];
    const int*   src = ei;
    const int*   dst = ei + EE;

    // per-layer params: ln_g, ln_b, Ws, bs, Wb, bb, bg  starting at index 2
    const float* P[3][7];
    for (int l = 0; l < 3; ++l)
        for (int j = 0; j < 7; ++j)
            P[l][j] = (const float*)d_in[2 + l * 7 + j];

    // workspace carve-out
    char* ws = (char*)d_ws;
    size_t off = 0;
    auto alloc = [&](size_t bytes) { void* p = ws + off; off += (bytes + 255) & ~(size_t)255; return p; };
    float*    dinv = (float*)   alloc((size_t)NN * 4);
    float*    pre  = (float*)   alloc((size_t)NN * 128 * 4);
    float*    h1   = (float*)   alloc((size_t)NN * 128 * 4);
    float*    h2   = (float*)   alloc((size_t)NN * 128 * 4);
    _Float16* Wt0  = (_Float16*)alloc((size_t)(128 * 5) * 128 * 2);
    _Float16* Wt1  = (_Float16*)alloc((size_t)(128 * 5) * 128 * 2);
    _Float16* Wt2  = (_Float16*)alloc((size_t)(384 * 5) * 64 * 2);
    float*    bc0  = (float*)   alloc(128 * 4);
    float*    bc1  = (float*)   alloc(128 * 4);
    float*    bc2  = (float*)   alloc(64 * 4);

    const int nblkN = (NN + 255) / 256;
    const int nblkE = (EE + 255) / 256;
    const int eWave = EE / 8;                 // wave-per-edge, 8 waves/block

    // 1) normalization coefficients
    deg_init  <<<nblkN, 256, 0, stream>>>(dinv);
    deg_acc   <<<nblkE, 256, 0, stream>>>(dst, dinv);
    deg_finish<<<nblkN, 256, 0, stream>>>(dinv);

    // 2) f16 weight packing (fragment-order layout)
    prep_weights<128, 128><<<(128 * 5 * 128 + 255) / 256, 256, 0, stream>>>(
        P[0][2], P[0][4], P[0][3], P[0][5], Wt0, bc0, HH);
    prep_weights<128, 128><<<(128 * 5 * 128 + 255) / 256, 256, 0, stream>>>(
        P[1][2], P[1][4], P[1][3], P[1][5], Wt1, bc1, HH);
    prep_weights<384, 64><<<(384 * 5 * 64 + 255) / 256, 256, 0, stream>>>(
        P[2][2], P[2][4], P[2][3], P[2][5], Wt2, bc2, CC);

    // 3) layer 0: FKAN(x) -> pre ; aggregate -> h1
    fkan_kernel<128, 0, 0, 128><<<NN / 16, 128, 0, stream>>>(
        x, x, x, P[0][0], P[0][1], Wt0, bc0, pre, HH);
    agg_init <<<(NN * HH + 255) / 256, 256, 0, stream>>>(pre, dinv, P[0][6], h1, HH);
    agg_edges<<<eWave, 256, 0, stream>>>(src, dst, dinv, pre, h1, HH);

    // 4) layer 1: FKAN(h1) -> pre ; aggregate -> h2
    fkan_kernel<128, 0, 0, 128><<<NN / 16, 128, 0, stream>>>(
        h1, h1, h1, P[1][0], P[1][1], Wt1, bc1, pre, HH);
    agg_init <<<(NN * HH + 255) / 256, 256, 0, stream>>>(pre, dinv, P[1][6], h2, HH);
    agg_edges<<<eWave, 256, 0, stream>>>(src, dst, dinv, pre, h2, HH);

    // 5) layer 2: FKAN([x|h1|h2]) -> pre(47) ; aggregate -> d_out
    fkan_kernel<128, 128, 128, 64><<<NN / 16, 128, 0, stream>>>(
        x, h1, h2, P[2][0], P[2][1], Wt2, bc2, pre, CC);
    float* out = (float*)d_out;
    agg_init <<<(NN * CC + 255) / 256, 256, 0, stream>>>(pre, dinv, P[2][6], out, CC);
    agg_edges<<<eWave, 256, 0, stream>>>(src, dst, dinv, pre, out, CC);
}
